// GGNN_31147102831270
// MI455X (gfx1250) — matrix operand order repaired and checked by reading, not compiled, not run on hardware
//
#include <hip/hip_runtime.h>
#include <hip/hip_bf16.h>
#include <stdint.h>

#define CDIM 128
#define GDIM 64

typedef __attribute__((ext_vector_type(16))) __bf16  v16bf;
typedef __attribute__((ext_vector_type(8)))  __bf16  v8bf;
typedef __attribute__((ext_vector_type(8)))  float   v8f;
typedef __attribute__((ext_vector_type(4)))  unsigned int   u32x4;
typedef __attribute__((ext_vector_type(4)))  unsigned short u16x4;
typedef __attribute__((ext_vector_type(4)))  float   f32x4;

// ---------- bf16 helpers (bit-exact, no header dependence) ----------
__device__ __forceinline__ unsigned short f2bf_bits(float f) {
    union { float f; unsigned u; } v; v.f = f;
    unsigned r = (v.u + 0x7FFFu + ((v.u >> 16) & 1u)) >> 16;   // RNE
    return (unsigned short)r;
}
__device__ __forceinline__ float bf2f(unsigned short s) {
    union { unsigned u; float f; } v; v.u = ((unsigned)s) << 16;
    return v.f;
}
__device__ __forceinline__ __bf16 us_as_bf(unsigned short s) {
    union { unsigned short s; __bf16 b; } u; u.s = s; return u.b;
}

// ---------- WMMA fragment loaders (CDNA5 16x16x32 bf16 layouts) ----------
// A 16x32: lane<16 -> row=lane, K = {k0+0..7, k0+16..23}; lane>=16 -> row=lane-16, K = {k0+8..15, k0+24..31}
__device__ __forceinline__ v16bf load_a_bf(const unsigned short* rowp, int k0, int hsel) {
    union { u32x4 u; v8bf b; } lo, hi;
    lo.u = *(const u32x4*)(rowp + k0 + hsel * 8);
    hi.u = *(const u32x4*)(rowp + k0 + 16 + hsel * 8);
    v16bf a;
#pragma unroll
    for (int j = 0; j < 8; ++j) { a[j] = lo.b[j]; a[j + 8] = hi.b[j]; }
    return a;
}
// A from fp32 source (convert on the fly)
__device__ __forceinline__ v16bf load_a_f32(const float* rowp, int k0, int hsel) {
    const float* p = rowp + k0 + hsel * 8;
    const float* q = rowp + k0 + 16 + hsel * 8;
    v16bf a;
#pragma unroll
    for (int j = 0; j < 8; ++j) {
        a[j]     = us_as_bf(f2bf_bits(p[j]));
        a[j + 8] = us_as_bf(f2bf_bits(q[j]));
    }
    return a;
}
// B 32x16: lane holds column lane&15; lane<16 -> K = {k0+0..7, k0+16..23}; lane>=16 -> K = {k0+8..15, k0+24..31}
__device__ __forceinline__ v16bf load_b_bf(const unsigned short* btrowp, int k0, int hsel) {
    union { u32x4 u; v8bf b; } lo, hi;
    lo.u = *(const u32x4*)(btrowp + k0 + hsel * 8);
    hi.u = *(const u32x4*)(btrowp + k0 + 16 + hsel * 8);
    v16bf b;
#pragma unroll
    for (int j = 0; j < 8; ++j) { b[j] = lo.b[j]; b[j + 8] = hi.b[j]; }
    return b;
}
__device__ __forceinline__ v8f wmma_bf16(v16bf a, v16bf b, v8f c) {
    return __builtin_amdgcn_wmma_f32_16x16x32_bf16(false, a, false, b, (short)0, c, false, false);
}

// ---------- prep kernels ----------
__global__ void k_init_h(const float* __restrict__ x, float* __restrict__ h,
                         unsigned short* __restrict__ hbf, int n) {
    int i = blockIdx.x * blockDim.x + threadIdx.x;
    if (i < n) { float v = x[i]; h[i] = v; hbf[i] = f2bf_bits(v); }
}
__global__ void k_f2bf(const float* __restrict__ s, unsigned short* __restrict__ d, int n) {
    int i = blockIdx.x * blockDim.x + threadIdx.x;
    if (i < n) d[i] = f2bf_bits(s[i]);
}
// wt[l][n][k] = w[l][k][n]  (so B^T rows are contiguous)
__global__ void k_transpose_w(const float* __restrict__ w, unsigned short* __restrict__ wt, int nmat) {
    int i = blockIdx.x * blockDim.x + threadIdx.x;
    int total = nmat * CDIM * CDIM;
    if (i < total) {
        int l = i / (CDIM * CDIM), r = i % (CDIM * CDIM);
        int n = r / CDIM, k = r % CDIM;
        wt[i] = f2bf_bits(w[l * CDIM * CDIM + k * CDIM + n]);
    }
}
__global__ void k_zero_f(float* __restrict__ p, int n) {
    int i = blockIdx.x * blockDim.x + threadIdx.x;
    if (i < n) p[i] = 0.0f;
}

// ---------- m = h @ W_l   (bf16 WMMA, bf16 output) ----------
__global__ void __launch_bounds__(256)
k_gemm_m(const unsigned short* __restrict__ hbf, const unsigned short* __restrict__ wt,
         unsigned short* __restrict__ mbf, int nnodes) {
    int wave = threadIdx.x >> 5, lane = threadIdx.x & 31;
    int r = lane & 15, hsel = lane >> 4;
    int row0 = blockIdx.x * 16;
    int arow = min(row0 + r, nnodes - 1);
    const unsigned short* ap = hbf + (size_t)arow * CDIM;
    const unsigned short* bp = wt + (size_t)(wave * 16 + r) * CDIM;
    v8f acc = {};
#pragma unroll
    for (int k0 = 0; k0 < CDIM; k0 += 32)
        acc = wmma_bf16(load_a_bf(ap, k0, hsel), load_b_bf(bp, k0, hsel), acc);
    int col = wave * 16 + r;
    unsigned short* op = mbf + (size_t)(row0 + hsel * 8) * CDIM + col;
    if (row0 + 16 <= nnodes) {             // full tile: unguarded, branch-free stores
#pragma unroll
        for (int i = 0; i < 8; ++i) op[(size_t)i * CDIM] = f2bf_bits(acc[i]);
    } else {                               // tail tile
#pragma unroll
        for (int i = 0; i < 8; ++i)
            if (row0 + hsel * 8 + i < nnodes) op[(size_t)i * CDIM] = f2bf_bits(acc[i]);
    }
}

// ---------- agg[dst] += m[src]  (wave per edge, f32 atomics) ----------
__global__ void __launch_bounds__(256)
k_scatter(const int* __restrict__ src, const int* __restrict__ dst,
          const unsigned short* __restrict__ mbf, float* __restrict__ agg, int E) {
    int gid = blockIdx.x * blockDim.x + threadIdx.x;
    int e = gid >> 5, lane = gid & 31;
    if (e >= E) return;
    int s = src[e], d = dst[e];
    u16x4 mv = *(const u16x4*)(mbf + (size_t)s * CDIM + lane * 4);
    float* ap = agg + (size_t)d * CDIM + lane * 4;
    atomicAdd(ap + 0, bf2f(mv.x));
    atomicAdd(ap + 1, bf2f(mv.y));
    atomicAdd(ap + 2, bf2f(mv.z));
    atomicAdd(ap + 3, bf2f(mv.w));
}

// ---------- fused GRU: gi = agg@Wih^T+bih, gh = h@Whh^T+bhh, elementwise update ----------
__global__ void __launch_bounds__(256)
k_gru(const float* __restrict__ agg,
      const unsigned short* __restrict__ wihbf, const unsigned short* __restrict__ whhbf,
      const float* __restrict__ bih, const float* __restrict__ bhh,
      float* __restrict__ h, unsigned short* __restrict__ hbf, int nnodes) {
    int wave = threadIdx.x >> 5, lane = threadIdx.x & 31;
    int r = lane & 15, hsel = lane >> 4;
    int row0 = blockIdx.x * 16;
    int arow = min(row0 + r, nnodes - 1);
    const float*          aggp = agg + (size_t)arow * CDIM;
    const unsigned short* hp   = hbf + (size_t)arow * CDIM;
    int colw = wave * 16 + r;   // output column for this lane (within each third)
    const unsigned short* bpir = wihbf + (size_t)(0 * CDIM + colw) * CDIM;
    const unsigned short* bpiz = wihbf + (size_t)(1 * CDIM + colw) * CDIM;
    const unsigned short* bpin = wihbf + (size_t)(2 * CDIM + colw) * CDIM;
    const unsigned short* bphr = whhbf + (size_t)(0 * CDIM + colw) * CDIM;
    const unsigned short* bphz = whhbf + (size_t)(1 * CDIM + colw) * CDIM;
    const unsigned short* bphn = whhbf + (size_t)(2 * CDIM + colw) * CDIM;

    v8f gir = {}, giz = {}, gin = {}, ghr = {}, ghz = {}, ghn = {};
#pragma unroll
    for (int k0 = 0; k0 < CDIM; k0 += 32) {
        v16bf Aa = load_a_f32(aggp, k0, hsel);
        v16bf Ah = load_a_bf(hp, k0, hsel);
        gir = wmma_bf16(Aa, load_b_bf(bpir, k0, hsel), gir);
        giz = wmma_bf16(Aa, load_b_bf(bpiz, k0, hsel), giz);
        gin = wmma_bf16(Aa, load_b_bf(bpin, k0, hsel), gin);
        ghr = wmma_bf16(Ah, load_b_bf(bphr, k0, hsel), ghr);
        ghz = wmma_bf16(Ah, load_b_bf(bphz, k0, hsel), ghz);
        ghn = wmma_bf16(Ah, load_b_bf(bphn, k0, hsel), ghn);
    }
    float vbir = bih[colw], vbiz = bih[CDIM + colw], vbin = bih[2 * CDIM + colw];
    float vbhr = bhh[colw], vbhz = bhh[CDIM + colw], vbhn = bhh[2 * CDIM + colw];

    __syncthreads();   // all waves done reading h/hbf A-fragments before in-place update

    float*          hopf = h   + (size_t)(row0 + hsel * 8) * CDIM + colw;
    unsigned short* hopb = hbf + (size_t)(row0 + hsel * 8) * CDIM + colw;
    bool full = (row0 + 16 <= nnodes);
#pragma unroll
    for (int i = 0; i < 8; ++i) {
        if (full || (row0 + hsel * 8 + i < nnodes)) {
            float rr = 1.0f / (1.0f + expf(-(gir[i] + vbir + ghr[i] + vbhr)));
            float zz = 1.0f / (1.0f + expf(-(giz[i] + vbiz + ghz[i] + vbhz)));
            float nn = tanhf(gin[i] + vbin + rr * (ghn[i] + vbhn));
            float hold = hopf[(size_t)i * CDIM];
            float hnew = (1.0f - zz) * nn + zz * hold;
            hopf[(size_t)i * CDIM] = hnew;
            hopb[(size_t)i * CDIM] = f2bf_bits(hnew);
        }
    }
}

// ---------- gate = h @ gate_w + gate_b  (wave per node) ----------
__global__ void __launch_bounds__(256)
k_gate(const float* __restrict__ h, const float* __restrict__ gw, const float* __restrict__ gb,
       float* __restrict__ gate, int nnodes) {
    int gid = blockIdx.x * blockDim.x + threadIdx.x;
    int node = gid >> 5, lane = gid & 31;
    if (node >= nnodes) return;
    f32x4 hv = *(const f32x4*)(h + (size_t)node * CDIM + lane * 4);
    f32x4 wv = *(const f32x4*)(gw + lane * 4);
    float s = hv.x * wv.x + hv.y * wv.y + hv.z * wv.z + hv.w * wv.w;
#pragma unroll
    for (int o = 16; o > 0; o >>= 1) s += __shfl_xor(s, o, 32);
    if (lane == 0) gate[node] = s + gb[0];
}

__global__ void k_init_seg(float* __restrict__ gmax, float* __restrict__ denom, int g) {
    int i = blockIdx.x * blockDim.x + threadIdx.x;
    if (i < g) { gmax[i] = -3.402823466e38f; denom[i] = 0.0f; }
}
__global__ void k_gmax(const float* __restrict__ gate, const int* __restrict__ batch,
                       float* __restrict__ gmax, int nnodes) {
    int i = blockIdx.x * blockDim.x + threadIdx.x;
    if (i >= nnodes) return;
    float v = gate[i];
    unsigned* a = (unsigned*)(gmax + batch[i]);
    unsigned old = *a;
    while (__uint_as_float(old) < v) {
        unsigned assumed = old;
        old = atomicCAS(a, assumed, __float_as_uint(v));
        if (old == assumed) break;
    }
}
__global__ void k_exp(const float* __restrict__ gate, const int* __restrict__ batch,
                      const float* __restrict__ gmax, float* __restrict__ ex,
                      float* __restrict__ denom, int nnodes) {
    int i = blockIdx.x * blockDim.x + threadIdx.x;
    if (i >= nnodes) return;
    int b = batch[i];
    float e = expf(gate[i] - gmax[b]);
    ex[i] = e;
    atomicAdd(denom + b, e);
}

// ---------- fused: x2 = tanh(h@lin1^T + b); out2 += x2; out1 += alpha*h ----------
__global__ void __launch_bounds__(256)
k_out(const float* __restrict__ h, const unsigned short* __restrict__ hbf,
      const unsigned short* __restrict__ l1bf, const float* __restrict__ l1b,
      const int* __restrict__ batch, const float* __restrict__ ex, const float* __restrict__ denom,
      float* __restrict__ out1, float* __restrict__ out2, int nnodes) {
    int wave = threadIdx.x >> 5, lane = threadIdx.x & 31;
    int r = lane & 15, hsel = lane >> 4;
    int row0 = blockIdx.x * 16;
    int arow = min(row0 + r, nnodes - 1);
    const unsigned short* ap = hbf + (size_t)arow * CDIM;
    int col = wave * 16 + r;
    const unsigned short* bp = l1bf + (size_t)col * CDIM;   // lin1_w row-major == B^T rows
    v8f acc = {};
#pragma unroll
    for (int k0 = 0; k0 < CDIM; k0 += 32)
        acc = wmma_bf16(load_a_bf(ap, k0, hsel), load_b_bf(bp, k0, hsel), acc);
    float bias = l1b[col];
    bool full = (row0 + 16 <= nnodes);
#pragma unroll
    for (int i = 0; i < 8; ++i) {
        int row = row0 + hsel * 8 + i;
        if (full || (row < nnodes)) {
            int b = batch[row];
            float al = ex[row] / denom[b];
            atomicAdd(&out2[b * CDIM + col], tanhf(acc[i] + bias));
            atomicAdd(&out1[b * CDIM + col], al * h[(size_t)row * CDIM + col]);
        }
    }
}

__global__ void k_final(const float* __restrict__ out1, const float* __restrict__ out2,
                        float* __restrict__ out, int n) {
    int i = blockIdx.x * blockDim.x + threadIdx.x;
    if (i < n) out[i] = tanhf(out1[i] * out2[i]);
}

// ---------------------------------------------------------------------------
extern "C" void kernel_launch(void* const* d_in, const int* in_sizes, int n_in,
                              void* d_out, int out_size, void* d_ws, size_t ws_size,
                              hipStream_t stream) {
    const float* x      = (const float*)d_in[0];
    const int*   eidx   = (const int*)d_in[1];
    const int*   batch  = (const int*)d_in[2];
    const float* ggnn_w = (const float*)d_in[3];
    const float* w_ih   = (const float*)d_in[4];
    const float* w_hh   = (const float*)d_in[5];
    const float* b_ih   = (const float*)d_in[6];
    const float* b_hh   = (const float*)d_in[7];
    const float* gate_w = (const float*)d_in[8];
    const float* gate_b = (const float*)d_in[9];
    const float* lin1_w = (const float*)d_in[10];
    const float* lin1_b = (const float*)d_in[11];

    const int N = in_sizes[2];
    const int E = in_sizes[1] / 2;
    const int L = in_sizes[3] / (CDIM * CDIM);
    const int* src = eidx;
    const int* dst = eidx + E;

    // workspace carve-out (256B aligned)
    char* w = (char*)d_ws;
    auto carve = [&](size_t bytes) -> char* {
        char* p = w; w += (bytes + 255) & ~(size_t)255; return p;
    };
    float*          h     = (float*)         carve((size_t)N * CDIM * 4);
    unsigned short* hbf   = (unsigned short*)carve((size_t)N * CDIM * 2);
    unsigned short* mbf   = (unsigned short*)carve((size_t)N * CDIM * 2);
    float*          agg   = (float*)         carve((size_t)N * CDIM * 4);
    unsigned short* wtbf  = (unsigned short*)carve((size_t)L * CDIM * CDIM * 2);
    unsigned short* wihbf = (unsigned short*)carve((size_t)3 * CDIM * CDIM * 2);
    unsigned short* whhbf = (unsigned short*)carve((size_t)3 * CDIM * CDIM * 2);
    unsigned short* l1bf  = (unsigned short*)carve((size_t)CDIM * CDIM * 2);
    float*          gate  = (float*)         carve((size_t)N * 4);
    float*          ex    = (float*)         carve((size_t)N * 4);
    float*          gmax  = (float*)         carve((size_t)GDIM * 4);
    float*          denom = (float*)         carve((size_t)GDIM * 4);
    float*          out12 = (float*)         carve((size_t)2 * GDIM * CDIM * 4);
    float*          out1  = out12;
    float*          out2  = out12 + GDIM * CDIM;

    const int T = 256;
    const int ntiles = (N + 15) / 16;

    // weight prep
    k_transpose_w<<<(L * CDIM * CDIM + T - 1) / T, T, 0, stream>>>(ggnn_w, wtbf, L);
    k_f2bf<<<(3 * CDIM * CDIM + T - 1) / T, T, 0, stream>>>(w_ih, wihbf, 3 * CDIM * CDIM);
    k_f2bf<<<(3 * CDIM * CDIM + T - 1) / T, T, 0, stream>>>(w_hh, whhbf, 3 * CDIM * CDIM);
    k_f2bf<<<(CDIM * CDIM + T - 1) / T, T, 0, stream>>>(lin1_w, l1bf, CDIM * CDIM);
    k_init_h<<<(N * CDIM + T - 1) / T, T, 0, stream>>>(x, h, hbf, N * CDIM);

    // GGNN propagation
    for (int l = 0; l < L; ++l) {
        k_gemm_m<<<ntiles, T, 0, stream>>>(hbf, wtbf + (size_t)l * CDIM * CDIM, mbf, N);
        k_zero_f<<<(N * CDIM + T - 1) / T, T, 0, stream>>>(agg, N * CDIM);
        k_scatter<<<(E * 32 + T - 1) / T, T, 0, stream>>>(src, dst, mbf, agg, E);
        k_gru<<<ntiles, T, 0, stream>>>(agg, wihbf, whhbf, b_ih, b_hh, h, hbf, N);
    }

    // attentional aggregation + readout
    k_gate<<<(N * 32 + T - 1) / T, T, 0, stream>>>(h, gate_w, gate_b, gate, N);
    k_init_seg<<<1, GDIM, 0, stream>>>(gmax, denom, GDIM);
    k_zero_f<<<(2 * GDIM * CDIM + T - 1) / T, T, 0, stream>>>(out12, 2 * GDIM * CDIM);
    k_gmax<<<(N + T - 1) / T, T, 0, stream>>>(gate, batch, gmax, N);
    k_exp<<<(N + T - 1) / T, T, 0, stream>>>(gate, batch, gmax, ex, denom, N);
    k_out<<<ntiles, T, 0, stream>>>(h, hbf, l1bf, lin1_b, batch, ex, denom, out1, out2, N);
    k_final<<<(GDIM * CDIM + T - 1) / T, T, 0, stream>>>(out1, out2, (float*)d_out, GDIM * CDIM);
}
